// Net_25950192402497
// MI455X (gfx1250) — compile-verified, matrix operand order checked by reading, not run on hardware
//
#include <hip/hip_runtime.h>
#include <hip/hip_bf16.h>

typedef __attribute__((ext_vector_type(16))) __bf16 v16bf;
typedef __attribute__((ext_vector_type(4)))  __bf16 v4bf;
typedef __attribute__((ext_vector_type(8)))  float  v8f;

// ---------------------------------------------------------------------------
// Weight conversion: fp32 [K x 128] -> bf16 in WMMA B-fragment layout:
//   frag[(kc*128 + n)*32 + k32]  (kc = K-chunk of 32, k contiguous per column)
// msg weights zero-padded K 272 -> 288.
// ---------------------------------------------------------------------------
__global__ void convert_weights(const float* __restrict__ pw1, const float* __restrict__ mw1,
                                const float* __restrict__ pw2, const float* __restrict__ mw2,
                                __bf16* __restrict__ wb1p, __bf16* __restrict__ wb1m,
                                __bf16* __restrict__ wb2p, __bf16* __restrict__ wb2m) {
    int i = blockIdx.x * blockDim.x + threadIdx.x;   // over 288*128
    if (i >= 288 * 128) return;
    int k = i >> 7;        // 0..287
    int n = i & 127;
    int kc = k >> 5, k32 = k & 31;
    size_t o = ((size_t)(kc * 128 + n)) * 32 + k32;
    if (k < 128) {
        wb1p[o] = (__bf16)pw1[k * 128 + n];
        wb2p[o] = (__bf16)pw2[k * 128 + n];
    }
    float v1 = (k < 272) ? mw1[k * 128 + n] : 0.0f;
    float v2 = (k < 272) ? mw2[k * 128 + n] : 0.0f;
    wb1m[o] = (__bf16)v1;
    wb2m[o] = (__bf16)v2;
}

__global__ void zero_f32(float* __restrict__ p, long long n) {
    long long i = (long long)blockIdx.x * blockDim.x + threadIdx.x;
    if (i < n) p[i] = 0.0f;
}

// ---------------------------------------------------------------------------
// Fused GEMM:  out = A @ W + scale*bias   (bf16 WMMA, f32 accumulate)
//   DEG=0: A row = A0[row]                     (K_PAD = 128)
//   DEG=1: A row = [deg*A0 | A1 | A2 | zeros]  (K_PAD = 288), bias scaled by deg
// Block: 256 threads = 8 waves; block = 32 rows x 128 cols (two 16-row A
// tiles). Each wave owns one 16-col tile and reuses its B fragment for both
// row tiles -> 2 WMMAs per K-chunk per wave, B traffic halved.
// ---------------------------------------------------------------------------
template <int K_PAD, bool DEG, bool RELU>
__global__ __launch_bounds__(256) void gemm_msg(const float* __restrict__ A0,
                                                const float* __restrict__ A1,
                                                const float* __restrict__ A2,
                                                const float* __restrict__ degv,
                                                const __bf16* __restrict__ Wb,
                                                const float* __restrict__ bias,
                                                float* __restrict__ out,
                                                int n_rows) {
    __shared__ __bf16 lsA[32 * K_PAD];   // 32 rows, K-major
    __shared__ float  lsDeg[32];

    const int tid  = threadIdx.x;
    const int lane = tid & 31;
    const int wave = tid >> 5;           // 0..7 -> output column tile
    const int blockRow = blockIdx.x * 32;

    // --- stage deg tile (epilogue use) ---
    if (DEG && tid < 32) {
        int grow = blockRow + tid;
        int gr = (grow < n_rows) ? grow : (n_rows - 1);
        lsDeg[tid] = degv[gr];
    }

    // --- stage A tiles: float4 global reads -> packed 4x bf16 (8B LDS stores)
    constexpr int QUADS = 32 * K_PAD / 4;
    for (int i4 = tid; i4 < QUADS; i4 += 256) {
        int r  = i4 / (K_PAD / 4);
        int k  = (i4 - r * (K_PAD / 4)) * 4;
        int grow = blockRow + r;
        int gr = (grow < n_rows) ? grow : (n_rows - 1);
        float4 v;
        if (!DEG) {
            v = *(const float4*)&A0[(size_t)gr * 128 + k];
        } else if (k < 128) {
            v = *(const float4*)&A0[(size_t)gr * 128 + k];
            float d = degv[gr];
            v.x *= d; v.y *= d; v.z *= d; v.w *= d;
        } else if (k < 256) {
            v = *(const float4*)&A1[(size_t)gr * 128 + (k - 128)];
        } else if (k < 272) {
            v = *(const float4*)&A2[(size_t)gr * 16 + (k - 256)];
        } else {
            v = make_float4(0.f, 0.f, 0.f, 0.f);
        }
        v4bf p;
        p.x = (__bf16)v.x; p.y = (__bf16)v.y; p.z = (__bf16)v.z; p.w = (__bf16)v.w;
        *(v4bf*)&lsA[r * K_PAD + k] = p;
    }
    __syncthreads();

    const int row = lane & 15;
    const int kb  = (lane < 16) ? 0 : 8;   // ISA 16-bit fragment K base
    const int col = wave * 16 + row;

    v8f acc0 = {}, acc1 = {};
#pragma unroll
    for (int kc = 0; kc < K_PAD / 32; ++kc) {
        // B fragment straight from global pre-shuffled layout (L2-hot)
        v16bf b;
        const __bf16* pb = Wb + ((size_t)(kc * 128 + col)) * 32 + kb;
        ((uint4*)&b)[0] = *(const uint4*)(pb);
        ((uint4*)&b)[1] = *(const uint4*)(pb + 16);
        // A fragments (row tile 0 and 1): elems 0..7 = K kb..kb+7, 8..15 = +16
        v16bf a0, a1;
        const __bf16* pa0 = &lsA[row * K_PAD + kc * 32 + kb];
        ((uint4*)&a0)[0] = *(const uint4*)(pa0);
        ((uint4*)&a0)[1] = *(const uint4*)(pa0 + 16);
        const __bf16* pa1 = pa0 + 16 * K_PAD;
        ((uint4*)&a1)[0] = *(const uint4*)(pa1);
        ((uint4*)&a1)[1] = *(const uint4*)(pa1 + 16);

        acc0 = __builtin_amdgcn_wmma_f32_16x16x32_bf16(
            false, a0, false, b, (short)0, acc0, false, false);
        acc1 = __builtin_amdgcn_wmma_f32_16x16x32_bf16(
            false, a1, false, b, (short)0, acc1, false, false);
    }

    // --- epilogue: C layout -> row = r + (lane<16?0:8), col = wave*16+(lane&15)
    const int mb = (lane < 16) ? 0 : 8;
    const float biasv = bias[col];

    float ds0[8] = {1, 1, 1, 1, 1, 1, 1, 1};
    float ds1[8] = {1, 1, 1, 1, 1, 1, 1, 1};
    if (DEG) {
        float4 d0 = *(const float4*)&lsDeg[mb];
        float4 d1 = *(const float4*)&lsDeg[mb + 4];
        float4 d2 = *(const float4*)&lsDeg[16 + mb];
        float4 d3 = *(const float4*)&lsDeg[16 + mb + 4];
        ds0[0] = d0.x; ds0[1] = d0.y; ds0[2] = d0.z; ds0[3] = d0.w;
        ds0[4] = d1.x; ds0[5] = d1.y; ds0[6] = d1.z; ds0[7] = d1.w;
        ds1[0] = d2.x; ds1[1] = d2.y; ds1[2] = d2.z; ds1[3] = d2.w;
        ds1[4] = d3.x; ds1[5] = d3.y; ds1[6] = d3.z; ds1[7] = d3.w;
    }

    float* po0 = out + (size_t)(blockRow + mb) * 128 + col;
    float* po1 = po0 + (size_t)16 * 128;

    if (blockRow + 32 <= n_rows) {   // uniform fast path: no per-element guards
#pragma unroll
        for (int r = 0; r < 8; ++r) {
            float v0 = acc0[r] + ds0[r] * biasv;
            float v1 = acc1[r] + ds1[r] * biasv;
            if (RELU) { v0 = fmaxf(v0, 0.0f); v1 = fmaxf(v1, 0.0f); }
            po0[(size_t)r * 128] = v0;
            po1[(size_t)r * 128] = v1;
        }
    } else {                         // tail block: guarded
#pragma unroll
        for (int r = 0; r < 8; ++r) {
            int g0 = blockRow + mb + r;
            int g1 = g0 + 16;
            float v0 = acc0[r] + ds0[r] * biasv;
            float v1 = acc1[r] + ds1[r] * biasv;
            if (RELU) { v0 = fmaxf(v0, 0.0f); v1 = fmaxf(v1, 0.0f); }
            if (g0 < n_rows) po0[(size_t)r * 128] = v0;
            if (g1 < n_rows) po1[(size_t)r * 128] = v1;
        }
    }
}

// ---------------------------------------------------------------------------
// Edge scatter: one wave per edge (hardware f32 atomics).
// ---------------------------------------------------------------------------
__global__ __launch_bounds__(256) void scatter_edges(const float* __restrict__ h,
                                                     const int* __restrict__ ei,
                                                     const float* __restrict__ ef,
                                                     float* __restrict__ agg_h,
                                                     float* __restrict__ agg_ef,
                                                     float* __restrict__ deg,
                                                     int E, int computeDeg) {
    int gid  = blockIdx.x * 256 + threadIdx.x;
    int e    = gid >> 5;
    int lane = gid & 31;
    if (e >= E) return;
    int src = ei[e];
    int dst = ei[E + e];

    float4 v = *(const float4*)(h + (size_t)src * 128 + lane * 4);
    float* ad = agg_h + (size_t)dst * 128 + lane * 4;
    unsafeAtomicAdd(ad + 0, v.x);
    unsafeAtomicAdd(ad + 1, v.y);
    unsafeAtomicAdd(ad + 2, v.z);
    unsafeAtomicAdd(ad + 3, v.w);

    if (lane < 16)
        unsafeAtomicAdd(agg_ef + (size_t)dst * 16 + lane, ef[(size_t)e * 16 + lane]);
    if (computeDeg && lane == 0)
        unsafeAtomicAdd(deg + dst, 1.0f);
}

// ---------------------------------------------------------------------------
// Link head: pred[q] = concat(x2[i0], x2[i1]) @ lpW[256x2] + lpb
// ---------------------------------------------------------------------------
__global__ __launch_bounds__(256) void head_kernel(const float* __restrict__ x2,
                                                   const int* __restrict__ eli,
                                                   const float* __restrict__ lpW,
                                                   const float* __restrict__ lpb,
                                                   float* __restrict__ out, int Q) {
    int q = blockIdx.x * blockDim.x + threadIdx.x;
    if (q >= Q) return;
    int i0 = eli[q];
    int i1 = eli[Q + q];
    float a0 = lpb[0], a1 = lpb[1];
    const float* r0 = x2 + (size_t)i0 * 128;
    const float* r1 = x2 + (size_t)i1 * 128;
#pragma unroll 4
    for (int c = 0; c < 128; ++c) {
        float f = r0[c];
        a0 = fmaf(f, lpW[c * 2 + 0], a0);
        a1 = fmaf(f, lpW[c * 2 + 1], a1);
    }
#pragma unroll 4
    for (int c = 0; c < 128; ++c) {
        float f = r1[c];
        a0 = fmaf(f, lpW[(128 + c) * 2 + 0], a0);
        a1 = fmaf(f, lpW[(128 + c) * 2 + 1], a1);
    }
    out[(size_t)q * 2 + 0] = a0;
    out[(size_t)q * 2 + 1] = a1;
}

// ---------------------------------------------------------------------------
extern "C" void kernel_launch(void* const* d_in, const int* in_sizes, int n_in,
                              void* d_out, int out_size, void* d_ws, size_t ws_size,
                              hipStream_t stream) {
    const float* x        = (const float*)d_in[0];
    const int*   ei       = (const int*)d_in[1];
    const float* ef       = (const float*)d_in[2];
    const int*   eli      = (const int*)d_in[3];
    const float* c1_pre_W = (const float*)d_in[4];
    const float* c1_pre_b = (const float*)d_in[5];
    const float* c1_msg_W = (const float*)d_in[6];
    const float* c1_msg_b = (const float*)d_in[7];
    const float* c2_pre_W = (const float*)d_in[8];
    const float* c2_pre_b = (const float*)d_in[9];
    const float* c2_msg_W = (const float*)d_in[10];
    const float* c2_msg_b = (const float*)d_in[11];
    const float* lp_W     = (const float*)d_in[12];
    const float* lp_b     = (const float*)d_in[13];

    const int N = in_sizes[0] / 128;
    const int E = in_sizes[1] / 2;
    const int Q = in_sizes[3] / 2;

    // ---- carve workspace ----
    char* ws = (char*)d_ws;
    auto take = [&](size_t bytes) -> void* {
        void* p = (void*)ws;
        ws += (bytes + 255) & ~(size_t)255;
        return p;
    };
    __bf16* wb1p   = (__bf16*)take((size_t)128 * 128 * 2);
    __bf16* wb1m   = (__bf16*)take((size_t)288 * 128 * 2);
    __bf16* wb2p   = (__bf16*)take((size_t)128 * 128 * 2);
    __bf16* wb2m   = (__bf16*)take((size_t)288 * 128 * 2);
    float*  deg    = (float*)take((size_t)N * 4);
    float*  agg_ef = (float*)take((size_t)N * 16 * 4);
    float*  h      = (float*)take((size_t)N * 128 * 4);
    float*  agg_h  = (float*)take((size_t)N * 128 * 4);
    float*  x1     = (float*)take((size_t)N * 128 * 4);   // also holds x2

    const int gemmBlocks    = (N + 31) / 32;
    const int scatterBlocks = (E * 32 + 255) / 256;

    // 0. weights -> bf16, pre-shuffled into WMMA B-fragment layout
    convert_weights<<<(288 * 128 + 255) / 256, 256, 0, stream>>>(
        c1_pre_W, c1_msg_W, c2_pre_W, c2_msg_W, wb1p, wb1m, wb2p, wb2m);

    // 1. zero accumulators (deg only once; same graph for both convs)
    zero_f32<<<((long long)N * 128 + 255) / 256, 256, 0, stream>>>(agg_h, (long long)N * 128);
    zero_f32<<<((long long)N * 16 + 255) / 256, 256, 0, stream>>>(agg_ef, (long long)N * 16);
    zero_f32<<<(N + 255) / 256, 256, 0, stream>>>(deg, N);

    // ---- conv1 ----
    gemm_msg<128, false, true><<<gemmBlocks, 256, 0, stream>>>(
        x, nullptr, nullptr, nullptr, wb1p, c1_pre_b, h, N);
    scatter_edges<<<scatterBlocks, 256, 0, stream>>>(h, ei, ef, agg_h, agg_ef, deg, E, 1);
    gemm_msg<288, true, true><<<gemmBlocks, 256, 0, stream>>>(
        h, agg_h, agg_ef, deg, wb1m, c1_msg_b, x1, N);

    // ---- conv2 ----
    zero_f32<<<((long long)N * 128 + 255) / 256, 256, 0, stream>>>(agg_h, (long long)N * 128);
    zero_f32<<<((long long)N * 16 + 255) / 256, 256, 0, stream>>>(agg_ef, (long long)N * 16);
    gemm_msg<128, false, true><<<gemmBlocks, 256, 0, stream>>>(
        x1, nullptr, nullptr, nullptr, wb2p, c2_pre_b, h, N);
    scatter_edges<<<scatterBlocks, 256, 0, stream>>>(h, ei, ef, agg_h, agg_ef, deg, E, 0);
    gemm_msg<288, true, false><<<gemmBlocks, 256, 0, stream>>>(
        h, agg_h, agg_ef, deg, wb2m, c2_msg_b, x1, N);

    // ---- link-prediction head ----
    head_kernel<<<(Q + 255) / 256, 256, 0, stream>>>(x1, eli, lp_W, lp_b, (float*)d_out, Q);
}
